// VectorQuantizer_163208757313
// MI455X (gfx1250) — matrix operand order repaired and checked by reading, not compiled, not run on hardware
//
#include <hip/hip_runtime.h>

typedef __attribute__((ext_vector_type(16))) _Float16 v16h;
typedef __attribute__((ext_vector_type(8)))  _Float16 v8h;
typedef __attribute__((ext_vector_type(8)))  float    v8f;

#define N_TOK 65536   // B*H*W = 64*32*32
#define DDIM  256
#define KCB   1024

// argmin kernel tiling
#define CHUNK_COLS 64                      // codebook columns staged per chunk
#define NCHUNK     (KCB / CHUNK_COLS)      // 16
#define TILES_PER_CHUNK (CHUNK_COLS / 16)  // 4 per wave
#define CHUNK_HALFS (CHUNK_COLS * DDIM)    // 16384 f16 = 32 KB
#define ROWS_PER_BLOCK 256                 // 8 waves x 32 rows

// ---------------------------------------------------------------------------
// Kernel 1: [B, D, H, W] f32  ->  flat16 [N, D] f16 (token-major, d-contiguous)
// LDS-tiled 32x32 transpose so both global read and write are coalesced.
// ---------------------------------------------------------------------------
__global__ __launch_bounds__(256)
void vq_transpose_f16_kernel(const float* __restrict__ latents,
                             _Float16* __restrict__ flat16) {
  __shared__ float tile[32][33];
  const int b   = blockIdx.z;
  const int d0  = blockIdx.y * 32;
  const int hw0 = blockIdx.x * 32;
  const int tx  = threadIdx.x;   // 0..31
  const int ty  = threadIdx.y;   // 0..7
#pragma unroll
  for (int j = 0; j < 4; ++j) {
    const int d = d0 + ty + j * 8;
    tile[ty + j * 8][tx] =
        latents[((size_t)b * DDIM + d) * 1024 + hw0 + tx];
  }
  __syncthreads();
#pragma unroll
  for (int j = 0; j < 4; ++j) {
    const int hw = hw0 + ty + j * 8;
    const int n  = (b << 10) + hw;
    flat16[(size_t)n * DDIM + d0 + tx] = (_Float16)tile[tx][ty + j * 8];
  }
}

// ---------------------------------------------------------------------------
// Kernel 2: codebook f32 -> code16 [K, D] f16, plus exact f32 cnorm[k]=||c_k||^2
// ---------------------------------------------------------------------------
__global__ __launch_bounds__(256)
void vq_codebook_prep_kernel(const float* __restrict__ codebook,
                             _Float16* __restrict__ code16,
                             float* __restrict__ cnorm) {
  const int lane = threadIdx.x & 31;
  const int wave = threadIdx.x >> 5;
  const int k    = blockIdx.x * 8 + wave;
  float s = 0.0f;
#pragma unroll
  for (int j = 0; j < 8; ++j) {
    const int d   = lane + 32 * j;
    const float c = codebook[k * DDIM + d];
    code16[k * DDIM + d] = (_Float16)c;
    s += c * c;
  }
#pragma unroll
  for (int off = 16; off >= 1; off >>= 1) s += __shfl_xor(s, off, 32);
  if (lane == 0) cnorm[k] = s;
}

// ---------------------------------------------------------------------------
// Kernel 3: WMMA distance GEMM + running argmin.
//  - codebook staged global->LDS with double-buffered ASYNC copies
//  - each wave owns 32 rows (two register-resident A tiles), so every B
//    fragment read from LDS feeds TWO independent WMMA chains.
// score[m,n] = cnorm[n] - 2 * dot(flat16[m], code16[n])  (||x||^2 dropped)
// ---------------------------------------------------------------------------
__device__ __forceinline__ void async_stage_chunk(
    const _Float16* __restrict__ code16, _Float16* lbuf_sel,
    int chunk, int tid) {
  // chunk = CHUNK_COLS*DDIM f16 = 32 KB, contiguous in code16.
  // 256 threads x 128 B each, 8 x b128 async copies per thread.
  const unsigned goff_base = (unsigned)(chunk * (CHUNK_HALFS * 2) + tid * 128);
  const unsigned lds_base  = (unsigned)(uintptr_t)lbuf_sel + (unsigned)(tid * 128);
#pragma unroll
  for (int i = 0; i < 8; ++i) {
    const unsigned lo = lds_base + i * 16;
    const unsigned go = goff_base + i * 16;
    asm volatile("global_load_async_to_lds_b128 %0, %1, %2"
                 :
                 : "v"(lo), "v"(go), "s"(code16)
                 : "memory");
  }
}

__global__ __launch_bounds__(256)
void vq_argmin_kernel(const _Float16* __restrict__ flat16,
                      const _Float16* __restrict__ code16,
                      const float* __restrict__ cnorm,
                      int* __restrict__ idx) {
  __shared__ _Float16 lbuf[2][CHUNK_HALFS];  // 2 x 32 KB double buffer

  const int tid     = threadIdx.x;
  const int lane    = tid & 31;
  const int wave    = tid >> 5;
  const int rowbase = blockIdx.x * ROWS_PER_BLOCK + wave * 32;
  const int m       = lane & 15;   // A row / B column within tile
  const int hl      = lane >> 4;   // lane half selects k-chunks

  // --- A fragments (two 16-row tiles, all of K), contiguous k order:
  // half 0: K 0..15 ; half 1: K 16..31
  v16h a0[8], a1[8];
  {
    const _Float16* arow0 = flat16 + (size_t)(rowbase + m) * DDIM;
    const _Float16* arow1 = arow0 + (size_t)16 * DDIM;
#pragma unroll
    for (int kt = 0; kt < 8; ++kt) {
      const int k0 = kt * 32;
      v8h lo0 = *(const v8h*)(arow0 + k0 + hl * 16);
      v8h hi0 = *(const v8h*)(arow0 + k0 + 8 + hl * 16);
      v8h lo1 = *(const v8h*)(arow1 + k0 + hl * 16);
      v8h hi1 = *(const v8h*)(arow1 + k0 + 8 + hl * 16);
#pragma unroll
      for (int i = 0; i < 8; ++i) {
        a0[kt][i] = lo0[i]; a0[kt][8 + i] = hi0[i];
        a1[kt][i] = lo1[i]; a1[kt][8 + i] = hi1[i];
      }
    }
  }

  float minv0[8], minv1[8];
  int   mini0[8], mini1[8];
#pragma unroll
  for (int r = 0; r < 8; ++r) {
    minv0[r] = 3.0e38f; mini0[r] = 0;
    minv1[r] = 3.0e38f; mini1[r] = 0;
  }

  // Prime the pipeline: chunk 0 -> buffer 0.
  async_stage_chunk(code16, &lbuf[0][0], 0, tid);

  for (int ch = 0; ch < NCHUNK; ++ch) {
    const int sel = ch & 1;
    // Issue next chunk's async copies into the other buffer (overlaps compute).
    if (ch + 1 < NCHUNK) {
      async_stage_chunk(code16, &lbuf[sel ^ 1][0], ch + 1, tid);
      // 8 newer async ops outstanding; <=8 means current chunk has landed.
      asm volatile("s_wait_asynccnt 8" ::: "memory");
    } else {
      asm volatile("s_wait_asynccnt 0" ::: "memory");
    }
    __syncthreads();

#pragma unroll
    for (int j = 0; j < TILES_PER_CHUNK; ++j) {
      const int n = (ch * TILES_PER_CHUNK + j) * 16 + m;
      // B column n = staged codebook row (d-contiguous in LDS);
      // half selects K 0..15 / 16..31 of each 32-wide k-step.
      const _Float16* bcol = &lbuf[sel][(j * 16 + m) * DDIM + hl * 16];
      v8f c0 = {}, c1 = {};
#pragma unroll
      for (int kt = 0; kt < 8; ++kt) {
        v16h b = *(const v16h*)(bcol + kt * 32);       // ds_load_b128 x2
        c0 = __builtin_amdgcn_wmma_f32_16x16x32_f16(
                 false, a0[kt], false, b, (short)0, c0, false, false);
        c1 = __builtin_amdgcn_wmma_f32_16x16x32_f16(
                 false, a1[kt], false, b, (short)0, c1, false, false);
      }

      const float cn = cnorm[n];  // same n for every r in this lane
#pragma unroll
      for (int r = 0; r < 8; ++r) {
        const float d0 = cn - 2.0f * c0[r];  // rows rowbase + r + 8*hl
        const float d1 = cn - 2.0f * c1[r];  // rows rowbase + 16 + r + 8*hl
        if (d0 < minv0[r]) { minv0[r] = d0; mini0[r] = n; }
        if (d1 < minv1[r]) { minv1[r] = d1; mini1[r] = n; }
      }
    }
    __syncthreads();  // all waves done with lbuf[sel] before it is overwritten
  }

  // Reduce over the 16 lanes of each half (each half holds distinct rows).
#pragma unroll
  for (int off = 8; off >= 1; off >>= 1) {
#pragma unroll
    for (int r = 0; r < 8; ++r) {
      float ov = __shfl_xor(minv0[r], off, 32);
      int   oi = __shfl_xor(mini0[r], off, 32);
      if (ov < minv0[r] || (ov == minv0[r] && oi < mini0[r])) {
        minv0[r] = ov; mini0[r] = oi;
      }
      ov = __shfl_xor(minv1[r], off, 32);
      oi = __shfl_xor(mini1[r], off, 32);
      if (ov < minv1[r] || (ov == minv1[r] && oi < mini1[r])) {
        minv1[r] = ov; mini1[r] = oi;
      }
    }
  }
  if (m == 0) {
#pragma unroll
    for (int r = 0; r < 8; ++r) {
      idx[rowbase + r + 8 * hl]      = mini0[r];
      idx[rowbase + 16 + r + 8 * hl] = mini1[r];
    }
  }
}

// ---------------------------------------------------------------------------
// Kernel 4: exact f32 gather back to [B, D, H, W], out = x + (q - x),
// per-block partial sum of (q-x)^2 (deterministic, no atomics).
// ---------------------------------------------------------------------------
__global__ __launch_bounds__(256)
void vq_gather_loss_kernel(const float* __restrict__ latents,
                           const float* __restrict__ codebook,
                           const int* __restrict__ idx,
                           float* __restrict__ out,
                           float* __restrict__ partials) {
  const int i  = blockIdx.x * 256 + threadIdx.x;   // index into [B, D, H*W]
  const int d  = (i >> 10) & 255;
  const int hw = i & 1023;
  const int b  = i >> 18;
  const int n  = (b << 10) | hw;
  const float q = codebook[idx[n] * DDIM + d];
  const float x = latents[i];
  out[i] = x + (q - x);
  const float diff = q - x;
  float s = diff * diff;
#pragma unroll
  for (int off = 16; off >= 1; off >>= 1) s += __shfl_xor(s, off, 32);
  __shared__ float wsum[8];
  if ((threadIdx.x & 31) == 0) wsum[threadIdx.x >> 5] = s;
  __syncthreads();
  if (threadIdx.x == 0) {
    float t = 0.0f;
#pragma unroll
    for (int w = 0; w < 8; ++w) t += wsum[w];
    partials[blockIdx.x] = t;
  }
}

// ---------------------------------------------------------------------------
// Kernel 5: fixed-order final reduction of 65536 partials -> scalar loss.
// vq_loss = embed + 0.25*commit = 1.25 * mean((q-x)^2)
// ---------------------------------------------------------------------------
__global__ __launch_bounds__(256)
void vq_loss_final_kernel(const float* __restrict__ partials,
                          float* __restrict__ loss_out) {
  float s = 0.0f;
  for (int t = threadIdx.x; t < 65536; t += 256) s += partials[t];
#pragma unroll
  for (int off = 16; off >= 1; off >>= 1) s += __shfl_xor(s, off, 32);
  __shared__ float wsum[8];
  if ((threadIdx.x & 31) == 0) wsum[threadIdx.x >> 5] = s;
  __syncthreads();
  if (threadIdx.x == 0) {
    float t = 0.0f;
#pragma unroll
    for (int w = 0; w < 8; ++w) t += wsum[w];
    *loss_out = 1.25f * t / ((float)N_TOK * (float)DDIM);
  }
}

extern "C" void kernel_launch(void* const* d_in, const int* in_sizes, int n_in,
                              void* d_out, int out_size, void* d_ws, size_t ws_size,
                              hipStream_t stream) {
  const float* latents  = (const float*)d_in[0];  // [64, 256, 32, 32]
  const float* codebook = (const float*)d_in[1];  // [1024, 256]
  float* out = (float*)d_out;                     // quantized + 1 loss element

  // Workspace layout (bytes), ~33 MB total.
  char* ws = (char*)d_ws;
  _Float16* flat16   = (_Float16*)(ws);                      // 33,554,432
  _Float16* code16   = (_Float16*)(ws + 33554432);           //    524,288
  float*    cnorm    = (float*)   (ws + 34078720);           //      4,096
  int*      idx      = (int*)     (ws + 34082816);           //    262,144
  float*    partials = (float*)   (ws + 34344960);           //    262,144

  dim3 tgrid(32, 8, 64), tblk(32, 8);
  vq_transpose_f16_kernel<<<tgrid, tblk, 0, stream>>>(latents, flat16);
  vq_codebook_prep_kernel<<<KCB / 8, 256, 0, stream>>>(codebook, code16, cnorm);
  vq_argmin_kernel<<<N_TOK / ROWS_PER_BLOCK, 256, 0, stream>>>(
      flat16, code16, cnorm, idx);
  vq_gather_loss_kernel<<<(N_TOK * DDIM) / 256, 256, 0, stream>>>(
      latents, codebook, idx, out, partials);
  vq_loss_final_kernel<<<1, 256, 0, stream>>>(partials, out + (out_size - 1));
}
